// GlobalAttentionPooling_71665824301246
// MI455X (gfx1250) — matrix operand order repaired and checked by reading, not compiled, not run on hardware
//
#include <hip/hip_runtime.h>
#include <hip/hip_bf16.h>
#include <math.h>

#define NN 1000000
#define DD 256
#define BB 4096

#define W1STR 264   // W1 LDS row stride (ushorts): 528B = 16B-aligned, bank rot 4
#define XSTR  40    // x-stage LDS row stride (ushorts): 80B, 16B-aligned
#define XSSZ  (128 * XSTR)

typedef __attribute__((ext_vector_type(16))) __bf16 v16bf;
typedef __attribute__((ext_vector_type(8)))  float  v8f;

union FragB16 { v16bf v; unsigned int u[8]; uint4 q[2]; };

// round-to-nearest-even f32 -> bf16, packed pair (a = low half, b = high half)
__device__ __forceinline__ unsigned int pack_bf16(float a, float b) {
  union { float f; unsigned int u; } ua, ub;
  ua.f = a; ub.f = b;
  unsigned int x = ua.u; x += 0x7FFFu + ((x >> 16) & 1u);
  unsigned int y = ub.u; y += 0x7FFFu + ((y >> 16) & 1u);
  return (x >> 16) | (y & 0xFFFF0000u);
}

// -------- Kernel 1: logits[n] = gelu(LN(x@W1^T + b1)) . W2 ----------------
// 256 threads = 8 waves; 128 rows/block; W1 in LDS as bf16; x-stage double-
// buffered; B-fragment loads software-pipelined one tile ahead of the WMMAs
// via sched_group_barrier so the wait becomes s_wait_dscnt <= 2.
__global__ __launch_bounds__(256) void k_logits(
    const float* __restrict__ x,  const float* __restrict__ W1,
    const float* __restrict__ b1, const float* __restrict__ ln1w,
    const float* __restrict__ ln1b, const float* __restrict__ W2,
    float* __restrict__ logits) {
  extern __shared__ char smem[];
  unsigned short* w1s = (unsigned short*)smem;     // [256][264] bf16
  unsigned short* xsb = w1s + 256 * W1STR;         // 2 x [128][40] bf16
  float* b1s  = (float*)(xsb + 2 * XSSZ);          // [256]
  float* l1ws = b1s  + 256;
  float* l1bs = l1ws + 256;
  float* w2s  = l1bs + 256;

  const int tid  = threadIdx.x;
  const int lane = tid & 31;
  const int wave = tid >> 5;
  const int rowBase = blockIdx.x * 128;

  b1s[tid]  = b1[tid];
  l1ws[tid] = ln1w[tid];
  l1bs[tid] = ln1b[tid];
  w2s[tid]  = W2[tid];

  // stage W1 (row-major [n][k]) into LDS as bf16
  const float4* W14 = (const float4*)W1;
  #pragma unroll 4
  for (int t = 0; t < 64; ++t) {
    int idx = t * 256 + tid;               // float4 index
    float4 wv = W14[idx];
    int base = idx << 2;
    int row = base >> 8, col = base & 255;
    uint2 pk; pk.x = pack_bf16(wv.x, wv.y); pk.y = pack_bf16(wv.z, wv.w);
    *(uint2*)&w1s[row * W1STR + col] = pk;
  }

  const int m0   = wave * 16;
  const int half = lane >> 4;
  const int ln16 = lane & 15;

  const float4* x4 = (const float4*)x;
  float4 xv[4];

  auto loadX = [&](int kk) {
    #pragma unroll
    for (int i = 0; i < 4; ++i) {
      int idx = i * 256 + tid;             // 0..1023
      int r = idx >> 3, c4 = idx & 7;
      int gr = rowBase + r;
      float4 v = {0.f, 0.f, 0.f, 0.f};
      if (gr < NN) v = x4[gr * 64 + kk * 8 + c4];
      xv[i] = v;
    }
  };
  auto storeX = [&](unsigned short* dst) {
    #pragma unroll
    for (int i = 0; i < 4; ++i) {
      int idx = i * 256 + tid;
      int r = idx >> 3, c4 = idx & 7;
      uint2 pk;
      pk.x = pack_bf16(xv[i].x, xv[i].y);
      pk.y = pack_bf16(xv[i].z, xv[i].w);
      *(uint2*)&dst[r * XSTR + c4 * 4] = pk;
    }
  };
  auto loadB = [&](FragB16& f, int t, int kk) {
    const unsigned short* p = &w1s[(t * 16 + ln16) * W1STR + kk * 32 + half * 8];
    f.q[0] = *(const uint4*)p;
    f.q[1] = *(const uint4*)(p + 16);
  };

  v8f acc[16];
  #pragma unroll
  for (int t = 0; t < 16; ++t) acc[t] = v8f{};

  // prologue: stage x-slab for kk = 0
  loadX(0);
  storeX(xsb);
  __syncthreads();

  #pragma unroll 1
  for (int kk = 0; kk < 8; ++kk) {
    unsigned short* cur = xsb + (kk & 1) * XSSZ;
    unsigned short* nxt = xsb + ((kk + 1) & 1) * XSSZ;

    if (kk < 7) loadX(kk + 1);   // global loads overlap the tile loop below

    // A fragment: 16x32 bf16 (two contiguous 16B groups per lane)
    FragB16 a;
    const unsigned short* pa = &cur[(m0 + ln16) * XSTR + half * 8];
    a.q[0] = *(const uint4*)pa;
    a.q[1] = *(const uint4*)(pa + 16);

    // double-buffered B fragments: load t+1 while WMMA consumes t
    FragB16 bb[2];
    loadB(bb[0], 0, kk);
    #pragma unroll
    for (int t = 0; t < 16; ++t) {
      if (t < 15) loadB(bb[(t + 1) & 1], t + 1, kk);
      acc[t] = __builtin_amdgcn_wmma_f32_16x16x32_bf16(
          false, a.v, false, bb[t & 1].v, (short)0, acc[t], false, false);
    }

#if defined(__has_builtin)
#if __has_builtin(__builtin_amdgcn_sched_group_barrier)
    // Pin the pipelined order: [A(2)+B0(2)+B1(2) DS reads], then
    // (WMMA_t, B_{t+2} DS reads) x14, then the last two WMMAs.
    // 0x100 = DS read, 0x008 = MFMA/WMMA.
    __builtin_amdgcn_sched_group_barrier(0x100, 6, 0);
    #pragma unroll
    for (int t = 0; t < 14; ++t) {
      __builtin_amdgcn_sched_group_barrier(0x008, 1, 0);
      __builtin_amdgcn_sched_group_barrier(0x100, 2, 0);
    }
    __builtin_amdgcn_sched_group_barrier(0x008, 1, 0);
    __builtin_amdgcn_sched_group_barrier(0x008, 1, 0);
#endif
#endif

    __syncthreads();             // all reads of `cur` complete
    if (kk < 7) {
      storeX(nxt);
      __syncthreads();           // publish next slab
    }
  }

  // LayerNorm stats per row: row (m0 + e + 8*half) lives in one 16-lane half
  // at accumulator element e; butterfly over 4 xor steps stays within a half.
  float s1[8], s2[8];
  #pragma unroll
  for (int e = 0; e < 8; ++e) { s1[e] = 0.f; s2[e] = 0.f; }
  #pragma unroll
  for (int t = 0; t < 16; ++t) {
    float bv = b1s[t * 16 + ln16];
    #pragma unroll
    for (int e = 0; e < 8; ++e) {
      float hv = acc[t][e] + bv;
      s1[e] += hv; s2[e] += hv * hv;
    }
  }
  #pragma unroll
  for (int e = 0; e < 8; ++e) {
    #pragma unroll
    for (int m = 1; m <= 8; m <<= 1) {
      s1[e] += __shfl_xor(s1[e], m, 32);
      s2[e] += __shfl_xor(s2[e], m, 32);
    }
  }
  float mu[8], rs[8], dotv[8];
  #pragma unroll
  for (int e = 0; e < 8; ++e) {
    mu[e] = s1[e] * (1.f / 256.f);
    float var = s2[e] * (1.f / 256.f) - mu[e] * mu[e];
    rs[e] = rsqrtf(var + 1e-5f);
    dotv[e] = 0.f;
  }
  // normalize -> exact GELU -> dot with W2
  #pragma unroll
  for (int t = 0; t < 16; ++t) {
    int n = t * 16 + ln16;
    float bv = b1s[n], lw = l1ws[n], lb = l1bs[n], w2v = w2s[n];
    #pragma unroll
    for (int e = 0; e < 8; ++e) {
      float hv = acc[t][e] + bv;
      float hn = (hv - mu[e]) * rs[e] * lw + lb;
      float g = 0.5f * hn * (1.f + erff(hn * 0.70710678118654752f));
      dotv[e] += g * w2v;
    }
  }
  #pragma unroll
  for (int e = 0; e < 8; ++e) {
    #pragma unroll
    for (int m = 1; m <= 8; m <<= 1) dotv[e] += __shfl_xor(dotv[e], m, 32);
    if (ln16 == e) {
      int gr = rowBase + m0 + e + half * 8;
      if (gr < NN) logits[gr] = dotv[e];
    }
  }
}

// -------- Kernel 2: segment starts via binary search (batch is sorted) ----
__global__ void k_bounds(const int* __restrict__ batch, int* __restrict__ segs) {
  int s = blockIdx.x * blockDim.x + threadIdx.x;
  if (s > BB) return;
  if (s == BB) { segs[BB] = NN; return; }
  int lo = 0, hi = NN;
  while (lo < hi) {
    int mid = (lo + hi) >> 1;
    if (batch[mid] < s) lo = mid + 1; else hi = mid;
  }
  segs[s] = lo;
}

// -------- Kernel 3: segment softmax, weighted sum of x, output LayerNorm --
__global__ __launch_bounds__(256) void k_pool(
    const float* __restrict__ x, const float* __restrict__ logits,
    const int* __restrict__ segs, const float* __restrict__ ln2w,
    const float* __restrict__ ln2b, float* __restrict__ out) {
  __shared__ float red[8];
  __shared__ float red2[8];
  __shared__ float coefs[256];
  __shared__ float xach[4][256];
  const int b = blockIdx.x;
  const int tid = threadIdx.x;
  const int s = segs[b], e = segs[b + 1];

  // segment max
  float mx = -3.402823466e+38f;
  for (int i = s + tid; i < e; i += 256) mx = fmaxf(mx, logits[i]);
  #pragma unroll
  for (int m = 16; m >= 1; m >>= 1) mx = fmaxf(mx, __shfl_xor(mx, m, 32));
  if ((tid & 31) == 0) red[tid >> 5] = mx;
  __syncthreads();
  mx = red[0];
  #pragma unroll
  for (int w = 1; w < 8; ++w) mx = fmaxf(mx, red[w]);
  __syncthreads();

  // sum of exp((w - max)/T), T = 0.4
  float sm = 0.f;
  for (int i = s + tid; i < e; i += 256) sm += expf((logits[i] - mx) * 2.5f);
  #pragma unroll
  for (int m = 16; m >= 1; m >>= 1) sm += __shfl_xor(sm, m, 32);
  if ((tid & 31) == 0) red[tid >> 5] = sm;
  __syncthreads();
  sm = 0.f;
  #pragma unroll
  for (int w = 0; w < 8; ++w) sm += red[w];
  float invSum = 1.f / (sm + 1e-6f);
  __syncthreads();

  // weighted accumulation with b128 loads: 4 node-subgroups x 64 lanes x float4
  const int sub = tid >> 6;      // which node within a group of 4
  const int ch4 = tid & 63;      // float4 channel index
  const float4* x4 = (const float4*)x;
  float4 acc4 = {0.f, 0.f, 0.f, 0.f};
  for (int base = s; base < e; base += 256) {
    int cnt = min(256, e - base);
    if (tid < cnt)
      coefs[tid] = expf((logits[base + tid] - mx) * 2.5f) * invSum;
    __syncthreads();
    for (int j = sub; j < cnt; j += 4) {
      float c = coefs[j];
      float4 xr = x4[(base + j) * 64 + ch4];
      acc4.x += xr.x * c; acc4.y += xr.y * c;
      acc4.z += xr.z * c; acc4.w += xr.w * c;
    }
    __syncthreads();
  }

  // gather per-channel totals (each channel split across 4 sub-threads)
  xach[sub][ch4 * 4 + 0] = acc4.x;
  xach[sub][ch4 * 4 + 1] = acc4.y;
  xach[sub][ch4 * 4 + 2] = acc4.z;
  xach[sub][ch4 * 4 + 3] = acc4.w;
  __syncthreads();
  float accd = xach[0][tid] + xach[1][tid] + xach[2][tid] + xach[3][tid];

  // output LayerNorm across D=256 (one value per thread)
  float v1 = accd, v2 = accd * accd;
  #pragma unroll
  for (int m = 16; m >= 1; m >>= 1) {
    v1 += __shfl_xor(v1, m, 32);
    v2 += __shfl_xor(v2, m, 32);
  }
  if ((tid & 31) == 0) { red[tid >> 5] = v1; red2[tid >> 5] = v2; }
  __syncthreads();
  float S1 = 0.f, S2 = 0.f;
  #pragma unroll
  for (int w = 0; w < 8; ++w) { S1 += red[w]; S2 += red2[w]; }
  float mu = S1 * (1.f / 256.f);
  float var = S2 * (1.f / 256.f) - mu * mu;
  float rsv = rsqrtf(var + 1e-5f);
  out[b * 256 + tid] = (accd - mu) * rsv * ln2w[tid] + ln2b[tid];
}

extern "C" void kernel_launch(void* const* d_in, const int* in_sizes, int n_in,
                              void* d_out, int out_size, void* d_ws, size_t ws_size,
                              hipStream_t stream) {
  const float* x    = (const float*)d_in[0];
  const int*   batch= (const int*)  d_in[1];
  const float* W1   = (const float*)d_in[2];
  const float* b1   = (const float*)d_in[3];
  const float* ln1w = (const float*)d_in[4];
  const float* ln1b = (const float*)d_in[5];
  const float* W2   = (const float*)d_in[6];
  const float* ln2w = (const float*)d_in[7];
  const float* ln2b = (const float*)d_in[8];
  float* out = (float*)d_out;

  float* logits = (float*)d_ws;                                  // N floats
  int* segs = (int*)((char*)d_ws + (size_t)NN * sizeof(float));  // B+1 ints

  const size_t smem1 = (size_t)(256 * W1STR + 2 * XSSZ) * sizeof(unsigned short)
                     + (size_t)4 * 256 * sizeof(float);          // 159744 B
  hipFuncSetAttribute((const void*)k_logits,
                      hipFuncAttributeMaxDynamicSharedMemorySize, (int)smem1);

  dim3 blk(256);
  k_logits<<<dim3((NN + 127) / 128), blk, smem1, stream>>>(
      x, W1, b1, ln1w, ln1b, W2, logits);
  k_bounds<<<dim3((BB + 1 + 255) / 256), blk, 0, stream>>>(batch, segs);
  k_pool<<<dim3(BB), blk, 0, stream>>>(x, logits, segs, ln2w, ln2b, out);
}
